// MultiHeadAttention_21414706938466
// MI455X (gfx1250) — matrix operand order repaired and checked by reading, not compiled, not run on hardware
//
#include <hip/hip_runtime.h>
#include <hip/hip_bf16.h>

// ---------------- problem constants (match reference) ----------------
#define E_   1024
#define H_   16
#define D_   64
#define B_   4
#define S_   2048
#define R_   8
#define MTOT (B_ * S_)            // 8192 rows of x
#define SCALING_ 2.0f             // 16.0 / 8
#define SCALE_   0.125f           // D^-0.5

typedef _Float16 half_t;
typedef __attribute__((ext_vector_type(16))) _Float16 v16h;
typedef __attribute__((ext_vector_type(8)))  _Float16 v8h;
typedef __attribute__((ext_vector_type(8)))  float    v8f;
typedef __attribute__((ext_vector_type(4)))  int      v4i;

#if __has_builtin(__builtin_amdgcn_global_load_async_to_lds_b128)
#define HAVE_ASYNC_LDS 1
typedef __attribute__((address_space(1))) v4i gv4i_t;   // global int4
typedef __attribute__((address_space(3))) v4i lv4i_t;   // LDS int4
#endif

static __device__ inline void wait_async_le1() {
#ifdef HAVE_ASYNC_LDS
#if __has_builtin(__builtin_amdgcn_s_wait_asynccnt)
  __builtin_amdgcn_s_wait_asynccnt(1);
#else
  asm volatile("s_wait_asynccnt 0x1" ::: "memory");
#endif
#endif
}
static __device__ inline void wait_async_le0() {
#ifdef HAVE_ASYNC_LDS
#if __has_builtin(__builtin_amdgcn_s_wait_asynccnt)
  __builtin_amdgcn_s_wait_asynccnt(0);
#else
  asm volatile("s_wait_asynccnt 0x0" ::: "memory");
#endif
#endif
}

// ---------------- WMMA fragment loaders (CDNA5 wave32 layouts) ----------------
// A (16x32 f16, MxK): lanes 0-15 -> M; halves 0..7 = K(8*hi..), halves 8..15 = K(16+8*hi..)
static __device__ inline v16h load_a_frag(const half_t* __restrict__ A, long ld) {
  int lane = threadIdx.x & 31;
  int m = lane & 15, hi = lane >> 4;
  const half_t* p = A + (long)m * ld + 8 * hi;
  v8h lo = *(const v8h*)p;
  v8h hh = *(const v8h*)(p + 16);
  v16h r;
#pragma unroll
  for (int i = 0; i < 8; ++i) { r[i] = lo[i]; r[i + 8] = hh[i]; }
  return r;
}

// B (32x16 f16, KxN) from row-major [K, N]: lanes 0-15 -> N; halves 0..7 = K(8*hi..), halves 8..15 = K(16+8*hi..)
static __device__ inline v16h load_b_frag(const half_t* __restrict__ Bp, long ld) {
  int lane = threadIdx.x & 31;
  int n = lane & 15, hi = lane >> 4;
  v16h r;
#pragma unroll
  for (int i = 0; i < 8; ++i) { r[i] = Bp[(long)(8 * hi + i) * ld + n]; r[i + 8] = Bp[(long)(16 + 8 * hi + i) * ld + n]; }
  return r;
}

// B frag from an LDS slab stored as [32 rows(K)][64 cols], subtile column base tc
static __device__ inline v16h load_b_frag_lds(const half_t* slab, int tc) {
  int lane = threadIdx.x & 31;
  int n = lane & 15, hi = lane >> 4;
  v16h r;
#pragma unroll
  for (int i = 0; i < 8; ++i) { r[i] = slab[(8 * hi + i) * 64 + tc + n]; r[i + 8] = slab[(16 + 8 * hi + i) * 64 + tc + n]; }
  return r;
}

static __device__ inline v8f wmma16(v16h a, v16h b, v8f c) {
  return __builtin_amdgcn_wmma_f32_16x16x32_f16(false, a, false, b, (short)0, c, false, false);
}

// stage one 32K x 64N B-slab (4 KB) into LDS: 8 waves x (32 lanes x 16B) = 4 KB
static __device__ inline void stage_slab(const half_t* __restrict__ WT, long ldn,
                                         int k0, int col0, half_t* slab) {
  int tid = threadIdx.x;                 // 0..255
  int w = tid >> 5, l = tid & 31;
  int row = 4 * w + (l >> 3);            // 0..31
  int seg = (l & 7) * 8;                 // half offset within 64-col row
  const half_t* g = WT + (long)(k0 + row) * ldn + col0 + seg;
  half_t* d = slab + row * 64 + seg;
#ifdef HAVE_ASYNC_LDS
  __builtin_amdgcn_global_load_async_to_lds_b128((gv4i_t*)g, (lv4i_t*)d, 0, 0);
#else
  *(v8h*)d = *(const v8h*)g;             // synchronous fallback
#endif
}

// double-buffered async-staged GEMM core: wave tile 32M x 64N, K = E_
static __device__ inline void gemm_core(const half_t* __restrict__ X,
                                        const half_t* __restrict__ WT, long ldn,
                                        int row0, int col0, half_t* sbuf,
                                        v8f acc[2][4]) {
  const int NSLAB = E_ / 32;             // 32 K-steps
  stage_slab(WT, ldn, 0, col0, sbuf);
  for (int kb = 0; kb < NSLAB; ++kb) {
    int k0 = kb * 32;
    half_t* cur = sbuf + (kb & 1) * (32 * 64);
    if (kb + 1 < NSLAB) {
      stage_slab(WT, ldn, k0 + 32, col0, sbuf + ((kb + 1) & 1) * (32 * 64));
      wait_async_le1();                  // our chunk of `cur` has landed
    } else {
      wait_async_le0();
    }
    __syncthreads();                     // everyone's chunk of `cur` has landed
    v16h a0 = load_a_frag(X + (long)row0 * E_ + k0, E_);
    v16h a1 = load_a_frag(X + (long)(row0 + 16) * E_ + k0, E_);
#pragma unroll
    for (int t = 0; t < 4; ++t) {
      v16h bf = load_b_frag_lds(cur, t * 16);
      acc[0][t] = wmma16(a0, bf, acc[0][t]);
      acc[1][t] = wmma16(a1, bf, acc[1][t]);
    }
    __syncthreads();                     // done reading before next overwrite
  }
}

// ---------------- tiny prep kernels ----------------
__global__ void k_cast_x(const float* __restrict__ x, half_t* __restrict__ x16, int n) {
  int i = blockIdx.x * blockDim.x + threadIdx.x;
  if (i < n) x16[i] = (half_t)x[i];
}

// W_effT[k][n] = W[n][k] + SCALING * sum_r Bm[n][r] * Am[r][k]   (f16, transposed)
__global__ void k_fold(const float* __restrict__ W, const float* __restrict__ Am,
                       const float* __restrict__ Bm, half_t* __restrict__ WT,
                       int N, int K) {
  int i = blockIdx.x * blockDim.x + threadIdx.x;
  if (i >= N * K) return;
  int nn = i / K, kk = i % K;
  float acc = W[(long)nn * K + kk];
  float s = 0.f;
#pragma unroll
  for (int r = 0; r < R_; ++r) s += Bm[nn * R_ + r] * Am[r * K + kk];
  WT[(long)kk * N + nn] = (half_t)(acc + SCALING_ * s);
}

// ---------------- GEMM1: qkv = x16 @ WqkvT, routed into Q / K^T / V ----------------
__global__ void __launch_bounds__(256) k_gemm_qkv(
    const half_t* __restrict__ X,   // [MTOT, E]
    const half_t* __restrict__ WT,  // [E, 3E]
    half_t* __restrict__ Qb,        // [B*H, S, D]
    half_t* __restrict__ KTb,       // [B*H, D, S]
    half_t* __restrict__ Vb) {      // [B*H, S, D]
  __shared__ __align__(128) half_t sbuf[2 * 32 * 64];
  const int NT = (3 * E_) / 64;     // 48 column tiles
  int nt = blockIdx.x % NT;
  int mtM = blockIdx.x / NT;        // 256-row macro tile
  int w = threadIdx.x >> 5;
  int row0 = mtM * 256 + w * 32;
  int col0 = nt * 64;

  v8f acc[2][4] = {};
  gemm_core(X, WT, 3 * E_, row0, col0, sbuf, acc);

  int lane = threadIdx.x & 31;
  int n = lane & 15, hi = lane >> 4;
  int bat = row0 / S_;              // constant across the 32-row wave tile
  int s0 = row0 % S_;
#pragma unroll
  for (int af = 0; af < 2; ++af) {
#pragma unroll
    for (int t = 0; t < 4; ++t) {
      int nc = col0 + t * 16;
      int c = nc / E_;
      int e0 = nc % E_;
      int h = e0 / D_;
      int d0 = e0 % D_;
      long bh = (long)bat * H_ + h;
      if (c == 1) {                 // K: transposed [D, S]; 8 contiguous along s
        v8h pk;
#pragma unroll
        for (int r = 0; r < 8; ++r) pk[r] = (half_t)acc[af][t][r];
        *(v8h*)(KTb + (bh * D_ + d0 + n) * (long)S_ + s0 + 16 * af + 8 * hi) = pk;
      } else {
        half_t* dst = (c == 0) ? Qb : Vb;
#pragma unroll
        for (int r = 0; r < 8; ++r) {
          int m = 16 * af + 8 * hi + r;
          dst[(bh * S_ + s0 + m) * (long)D_ + d0 + n] = (half_t)acc[af][t][r];
        }
      }
    }
  }
}

// ---------------- flash attention: one wave per (b, h, 16-row q tile) ----------------
__global__ void __launch_bounds__(32) k_flash(
    const half_t* __restrict__ Qb, const half_t* __restrict__ KTb,
    const half_t* __restrict__ Vb, half_t* __restrict__ O /* [MTOT, E] f16 */) {
  __shared__ __align__(32) float  sS[16][32];
  __shared__ __align__(32) half_t sP[16][32];
  __shared__ __align__(32) float  sCorr[16];
  __shared__ __align__(32) float  sLinv[16];

  const int QT = S_ / 16;
  int bh = blockIdx.x / QT;
  int qt = blockIdx.x % QT;
  int bat = bh / H_, h = bh % H_;
  int qr0 = qt * 16;
  int lane = threadIdx.x & 31;
  int n = lane & 15, hi = lane >> 4;

  const half_t* Qh = Qb + (long)bh * S_ * D_;
  const half_t* Kh = KTb + (long)bh * D_ * S_;
  const half_t* Vh = Vb + (long)bh * S_ * D_;

  v16h aq0 = load_a_frag(Qh + (long)qr0 * D_ + 0, D_);
  v16h aq1 = load_a_frag(Qh + (long)qr0 * D_ + 32, D_);

  v8f o0 = {}, o1 = {}, o2 = {}, o3 = {};
  float m_i = -3.0e38f, l_i = 0.0f;     // meaningful in lanes 0..15 only

  int kbmax = (qr0 + 15) / 32;          // causal horizon in 32-key blocks
  for (int kb = 0; kb <= kbmax; ++kb) {
    int k0 = kb * 32;
#pragma unroll
    for (int ntile = 0; ntile < 2; ++ntile) {
      int c0 = k0 + ntile * 16;
      v16h bk0 = load_b_frag(Kh + 0 * (long)S_ + c0, S_);
      v16h bk1 = load_b_frag(Kh + 32 * (long)S_ + c0, S_);
      v8f sc = {};
      sc = wmma16(aq0, bk0, sc);
      sc = wmma16(aq1, bk1, sc);
#pragma unroll
      for (int r = 0; r < 8; ++r) {
        int m = 8 * hi + r;
        int gq = qr0 + m, gk = c0 + n;
        sS[m][ntile * 16 + n] = (gk <= gq) ? sc[r] * SCALE_ : -1.0e30f;
      }
    }
    asm volatile("s_wait_dscnt 0x0" ::: "memory");
    if (lane < 16) {
      float bm = -3.0e38f;
#pragma unroll
      for (int c = 0; c < 32; ++c) bm = fmaxf(bm, sS[lane][c]);
      float nm = fmaxf(m_i, bm);
      float corr = __expf(m_i - nm);
      float sum = 0.0f;
#pragma unroll
      for (int c = 0; c < 32; ++c) {
        float p = __expf(sS[lane][c] - nm);
        sum += p;
        sP[lane][c] = (half_t)p;
      }
      l_i = l_i * corr + sum;
      m_i = nm;
      sCorr[lane] = corr;
    }
    asm volatile("s_wait_dscnt 0x0" ::: "memory");
    v8f cf = *(const v8f*)&sCorr[8 * hi];
#pragma unroll
    for (int r = 0; r < 8; ++r) { o0[r] *= cf[r]; o1[r] *= cf[r]; o2[r] *= cf[r]; o3[r] *= cf[r]; }
    {
      int m = lane & 15;
      v8h lo = *(const v8h*)&sP[m][8 * hi];
      v8h hhv = *(const v8h*)&sP[m][16 + 8 * hi];
      v16h pa;
#pragma unroll
      for (int i = 0; i < 8; ++i) { pa[i] = lo[i]; pa[i + 8] = hhv[i]; }
      v16h bv0 = load_b_frag(Vh + (long)k0 * D_ + 0, D_);
      v16h bv1 = load_b_frag(Vh + (long)k0 * D_ + 16, D_);
      v16h bv2 = load_b_frag(Vh + (long)k0 * D_ + 32, D_);
      v16h bv3 = load_b_frag(Vh + (long)k0 * D_ + 48, D_);
      o0 = wmma16(pa, bv0, o0);
      o1 = wmma16(pa, bv1, o1);
      o2 = wmma16(pa, bv2, o2);
      o3 = wmma16(pa, bv3, o3);
    }
  }
  if (lane < 16) sLinv[lane] = 1.0f / l_i;
  asm volatile("s_wait_dscnt 0x0" ::: "memory");
  v8f lf = *(const v8f*)&sLinv[8 * hi];
#pragma unroll
  for (int r = 0; r < 8; ++r) {
    int m = 8 * hi + r;
    long base = ((long)bat * S_ + qr0 + m) * E_ + h * D_ + n;
    O[base + 0]  = (half_t)(o0[r] * lf[r]);
    O[base + 16] = (half_t)(o1[r] * lf[r]);
    O[base + 32] = (half_t)(o2[r] * lf[r]);
    O[base + 48] = (half_t)(o3[r] * lf[r]);
  }
}

// ---------------- GEMM2: out = attn @ WprojT + bias (fp32 out) ----------------
__global__ void __launch_bounds__(256) k_gemm_proj(
    const half_t* __restrict__ Aa,  // [MTOT, E]
    const half_t* __restrict__ WT,  // [E, E]
    const float* __restrict__ bias, // [E]
    float* __restrict__ Out) {      // [MTOT, E]
  __shared__ __align__(128) half_t sbuf[2 * 32 * 64];
  const int NT = E_ / 64;           // 16 column tiles
  int nt = blockIdx.x % NT;
  int mtM = blockIdx.x / NT;
  int w = threadIdx.x >> 5;
  int row0 = mtM * 256 + w * 32;
  int col0 = nt * 64;

  v8f acc[2][4] = {};
  gemm_core(Aa, WT, E_, row0, col0, sbuf, acc);

  int lane = threadIdx.x & 31;
  int n = lane & 15, hi = lane >> 4;
#pragma unroll
  for (int af = 0; af < 2; ++af) {
#pragma unroll
    for (int t = 0; t < 4; ++t) {
      float bz = bias[col0 + t * 16 + n];
#pragma unroll
      for (int r = 0; r < 8; ++r) {
        int m = 16 * af + 8 * hi + r;
        Out[((long)row0 + m) * E_ + col0 + t * 16 + n] = acc[af][t][r] + bz;
      }
    }
  }
}

// ---------------- host-side launcher ----------------
extern "C" void kernel_launch(void* const* d_in, const int* in_sizes, int n_in,
                              void* d_out, int out_size, void* d_ws, size_t ws_size,
                              hipStream_t stream) {
  (void)in_sizes; (void)n_in; (void)out_size; (void)ws_size;
  const float* x     = (const float*)d_in[0];
  // d_in[1] = causal mask, implemented analytically
  const float* Wqkv  = (const float*)d_in[2];
  const float* Wproj = (const float*)d_in[3];
  const float* bproj = (const float*)d_in[4];
  const float* Aqkv  = (const float*)d_in[5];
  const float* Bqkv  = (const float*)d_in[6];
  const float* Aproj = (const float*)d_in[7];
  const float* Bproj = (const float*)d_in[8];
  float* out = (float*)d_out;

  char* ws = (char*)d_ws;
  size_t off = 0;
  half_t* x16 = (half_t*)(ws + off); off += (size_t)MTOT * E_ * 2;         // 16 MB
  half_t* WTq = (half_t*)(ws + off); off += (size_t)E_ * 3 * E_ * 2;       // 6 MB
  half_t* WTp = (half_t*)(ws + off); off += (size_t)E_ * E_ * 2;           // 2 MB
  half_t* Qb  = (half_t*)(ws + off); off += (size_t)B_ * H_ * S_ * D_ * 2; // 16 MB
  half_t* KTb = (half_t*)(ws + off); off += (size_t)B_ * H_ * S_ * D_ * 2; // 16 MB
  half_t* Vb  = (half_t*)(ws + off); off += (size_t)B_ * H_ * S_ * D_ * 2; // 16 MB
  half_t* At  = (half_t*)(ws + off); off += (size_t)MTOT * E_ * 2;         // 16 MB

  k_cast_x<<<(MTOT * E_ + 255) / 256, 256, 0, stream>>>(x, x16, MTOT * E_);
  k_fold<<<(3 * E_ * E_ + 255) / 256, 256, 0, stream>>>(Wqkv, Aqkv, Bqkv, WTq, 3 * E_, E_);
  k_fold<<<(E_ * E_ + 255) / 256, 256, 0, stream>>>(Wproj, Aproj, Bproj, WTp, E_, E_);

  // (MTOT/256) * 48 = 1536 macro-tile blocks
  k_gemm_qkv<<<(MTOT / 256) * ((3 * E_) / 64), 256, 0, stream>>>(x16, WTq, Qb, KTb, Vb);

  k_flash<<<B_ * H_ * (S_ / 16), 32, 0, stream>>>(Qb, KTb, Vb, At);

  // (MTOT/256) * 16 = 512 macro-tile blocks
  k_gemm_proj<<<(MTOT / 256) * (E_ / 64), 256, 0, stream>>>(At, WTp, bproj, out);
}
